// Attention_block_80616536146285
// MI455X (gfx1250) — matrix operand order repaired and checked by reading, not compiled, not run on hardware
//
#include <hip/hip_runtime.h>
#include <hip/hip_bf16.h>

// ---------------------------------------------------------------------------
// Attention gate block for MI455X (gfx1250), wave32, WMMA bf16.
// conv1x1 == GEMM [128 x 256] * [256 x 65536]; BN cancels conv biases exactly.
// Pipeline: pack W -> WMMA GEMM (G,X raw, bf16, L2-resident) -> per-channel
// stats -> fused BN+relu+psi dot -> psi BN+sigmoid -> gate multiply.
// ---------------------------------------------------------------------------

typedef __attribute__((ext_vector_type(16))) __bf16 bf16x16;
typedef __attribute__((ext_vector_type(2)))  __bf16 bf16x2;
typedef __attribute__((ext_vector_type(8)))  float  v8f;
typedef __attribute__((ext_vector_type(4)))  int    v4i;

#define F_INT   128
#define C_IN    256
#define HW      4096        // 64*64
#define NTOT    65536       // B*HW
#define MT      8           // M tiles of 16
#define KB      8           // K blocks of 32

#if defined(__gfx1250__) && __has_builtin(__builtin_amdgcn_global_load_async_to_lds_b128) && __has_builtin(__builtin_amdgcn_s_wait_asynccnt)
#define HAVE_ASYNC_LDS 1
#endif
#if defined(__gfx1250__) && __has_builtin(__builtin_amdgcn_cvt_pk_bf16_f32)
#define HAVE_PK_BF16 1
#endif

typedef __attribute__((address_space(1))) v4i as1_v4i;  // global
typedef __attribute__((address_space(3))) v4i as3_v4i;  // LDS

static __device__ __forceinline__ unsigned short f2bf(float f) {
    unsigned u = __float_as_uint(f);
    unsigned lsb = (u >> 16) & 1u;
    u += 0x7FFFu + lsb;                 // round-to-nearest-even
    return (unsigned short)(u >> 16);
}
static __device__ __forceinline__ float bf2f(unsigned short h) {
    return __uint_as_float(((unsigned)h) << 16);
}

// ---------------------------------------------------------------------------
// Workspace layout (bytes)
// ---------------------------------------------------------------------------
static constexpr size_t OFF_PWG   = 0;                       // 64 KiB
static constexpr size_t OFF_PWX   = 64 * 1024;               // 64 KiB
static constexpr size_t OFF_G     = 128 * 1024;              // 16 MiB bf16
static constexpr size_t SZ_GX     = (size_t)F_INT * NTOT * 2;
static constexpr size_t OFF_X     = OFF_G + SZ_GX;           // 16 MiB bf16
static constexpr size_t OFF_STATS = OFF_X + SZ_GX;           // 8 KiB floats
static constexpr size_t OFF_PRAW  = OFF_STATS + 8192;        // 256 KiB
static constexpr size_t OFF_GATE  = OFF_PRAW + (size_t)NTOT * 4;
// float indices inside stats region:
//   sumG@0 sqG@128 sumX@256 sqX@384 psum@512 psq@513
//   aG@640 bG@768 aX@896 bX@1024

// ---------------------------------------------------------------------------
__global__ void zero_stats(float* p, int n) {
    int i = blockIdx.x * blockDim.x + threadIdx.x;
    if (i < n) p[i] = 0.0f;
}

// ---------------------------------------------------------------------------
// 1) pack fp32 weights [128,256] into WMMA A-fragment order (bf16).
//    A 16-bit 16x32 layout (ISA 7.12.2).
// ---------------------------------------------------------------------------
__global__ void pack_weights(const float* __restrict__ w,
                             unsigned short* __restrict__ out) {
    int i = blockIdx.x * 256 + threadIdx.x;      // 32768 total
    if (i >= MT * KB * 32 * 16) return;
    int h    = i & 15;
    int lane = (i >> 4) & 31;
    int kb   = (i >> 9) & 7;
    int mt   = i >> 12;
    int v = h >> 1, pos = h & 1;
    int M = mt * 16 + (lane & 15);
    int K = kb * 32 + ((lane >> 4) << 3) + ((v >> 2) << 4) + ((v & 3) << 1) + pos;
    out[i] = f2bf(w[M * C_IN + K]);
}

// ---------------------------------------------------------------------------
// 2) GEMM: C[128 x 65536] = W * in, bf16 WMMA, fp32 accumulate.
//    512 blocks (16 batches x 32 col-blocks), 8 waves; wave owns 16 columns.
// ---------------------------------------------------------------------------
__global__ __launch_bounds__(256)
void conv_gemm(const float* __restrict__ in,           // [16,256,4096]
               const unsigned short* __restrict__ packedA,
               unsigned short* __restrict__ Cout) {    // bf16 [128][65536]
    __shared__ uint4 lAq[MT * KB * 32 * 2];            // 64 KiB: A fragments
    int tid  = threadIdx.x;
    int lane = tid & 31;
    int w    = tid >> 5;

    // stage packed weights into LDS (async tensor path when available)
#if HAVE_ASYNC_LDS
    {
        const char* gsrc = (const char*)packedA;
        char* ldst = (char*)lAq;
        for (int i = tid * 16; i < 64 * 1024; i += 256 * 16) {
            __builtin_amdgcn_global_load_async_to_lds_b128(
                (as1_v4i*)(gsrc + i), (as3_v4i*)(ldst + i), 0, 0);
        }
        __builtin_amdgcn_s_wait_asynccnt(0);
    }
#else
    {
        const uint4* src = (const uint4*)packedA;
        for (int i = tid; i < MT * KB * 32 * 2; i += 256) lAq[i] = src[i];
    }
#endif
    __syncthreads();

    int b   = blockIdx.x >> 5;
    int cb  = blockIdx.x & 31;
    int hw0 = cb * 128 + w * 16;
    const float* colBase = in + ((size_t)b * C_IN) * HW + hw0;

    v8f acc[MT] = {};
    const unsigned short* lA = (const unsigned short*)lAq;

    for (int kb = 0; kb < KB; ++kb) {
        // B fragment: lane L = column N (L%16), K rows 8*(L/16)+0..7 and 16+8*(L/16)+0..7
        const float* rp = colBase + (size_t)(kb * 32 + 8 * (lane >> 4)) * HW + (lane & 15);
        if (kb < KB - 1) __builtin_prefetch(rp + 32 * HW, 0, 1);
        float4 f0 = make_float4(rp[0 * HW], rp[1 * HW], rp[2 * HW], rp[3 * HW]);
        float4 f1 = make_float4(rp[4 * HW], rp[5 * HW], rp[6 * HW], rp[7 * HW]);
        float4 f2 = make_float4(rp[16 * HW], rp[17 * HW], rp[18 * HW], rp[19 * HW]);
        float4 f3 = make_float4(rp[20 * HW], rp[21 * HW], rp[22 * HW], rp[23 * HW]);
#if HAVE_PK_BF16
        union { bf16x2 p[8]; bf16x16 v; } ub;
        ub.p[0] = __builtin_amdgcn_cvt_pk_bf16_f32(f0.x, f0.y);
        ub.p[1] = __builtin_amdgcn_cvt_pk_bf16_f32(f0.z, f0.w);
        ub.p[2] = __builtin_amdgcn_cvt_pk_bf16_f32(f1.x, f1.y);
        ub.p[3] = __builtin_amdgcn_cvt_pk_bf16_f32(f1.z, f1.w);
        ub.p[4] = __builtin_amdgcn_cvt_pk_bf16_f32(f2.x, f2.y);
        ub.p[5] = __builtin_amdgcn_cvt_pk_bf16_f32(f2.z, f2.w);
        ub.p[6] = __builtin_amdgcn_cvt_pk_bf16_f32(f3.x, f3.y);
        ub.p[7] = __builtin_amdgcn_cvt_pk_bf16_f32(f3.z, f3.w);
#else
        union { unsigned short u[16]; bf16x16 v; } ub;
        ub.u[0]=f2bf(f0.x); ub.u[1]=f2bf(f0.y); ub.u[2]=f2bf(f0.z); ub.u[3]=f2bf(f0.w);
        ub.u[4]=f2bf(f1.x); ub.u[5]=f2bf(f1.y); ub.u[6]=f2bf(f1.z); ub.u[7]=f2bf(f1.w);
        ub.u[8]=f2bf(f2.x); ub.u[9]=f2bf(f2.y); ub.u[10]=f2bf(f2.z); ub.u[11]=f2bf(f2.w);
        ub.u[12]=f2bf(f3.x); ub.u[13]=f2bf(f3.y); ub.u[14]=f2bf(f3.z); ub.u[15]=f2bf(f3.w);
#endif
        // preload all 8 A fragments, then issue the 8 WMMAs back-to-back
        union AF { uint4 q[2]; bf16x16 v; } ua[MT];
        #pragma unroll
        for (int mt = 0; mt < MT; ++mt) {
            const uint4* ap = (const uint4*)(lA + ((mt * KB + kb) * 32 + lane) * 16);
            ua[mt].q[0] = ap[0];
            ua[mt].q[1] = ap[1];
        }
        #pragma unroll
        for (int mt = 0; mt < MT; ++mt) {
            acc[mt] = __builtin_amdgcn_wmma_f32_16x16x32_bf16(
                false, ua[mt].v, false, ub.v, (short)0, acc[mt], false, false);
        }
    }

    // store raw conv output (bf16); C/D layout: M = r + 8*(lane/16), N = lane%16
    int halfSel = lane >> 4;
    int nloc    = lane & 15;
    size_t ncol = (size_t)b * HW + hw0 + nloc;
    #pragma unroll
    for (int mt = 0; mt < MT; ++mt) {
        #pragma unroll
        for (int r = 0; r < 8; ++r) {
            int m = mt * 16 + r + 8 * halfSel;
            Cout[(size_t)m * NTOT + ncol] = f2bf(acc[mt][r]);
        }
    }
}

// ---------------------------------------------------------------------------
// 3) per-channel sum / sumsq over 65536 columns; one block per channel,
//    LDS tree reduction, direct store (no atomics, no pre-zeroing).
// ---------------------------------------------------------------------------
__global__ __launch_bounds__(256)
void channel_stats(const unsigned short* __restrict__ C,
                   float* __restrict__ sum, float* __restrict__ sq) {
    __shared__ float ls[256], lq[256];
    int tid = threadIdx.x;
    int c = blockIdx.x;
    const unsigned* row = (const unsigned*)(C + (size_t)c * NTOT);
    float s = 0.0f, q = 0.0f;
    for (int i = tid; i < NTOT / 2; i += 256) {
        unsigned u = row[i];
        float a = bf2f((unsigned short)(u & 0xffffu));
        float b = bf2f((unsigned short)(u >> 16));
        s += a + b;
        q = fmaf(a, a, q);
        q = fmaf(b, b, q);
    }
    ls[tid] = s; lq[tid] = q;
    __syncthreads();
    #pragma unroll
    for (int o = 128; o > 0; o >>= 1) {
        if (tid < o) { ls[tid] += ls[tid + o]; lq[tid] += lq[tid + o]; }
        __syncthreads();
    }
    if (tid == 0) { sum[c] = ls[0]; sq[c] = lq[0]; }
}

// ---------------------------------------------------------------------------
// 4) fold BN stats into per-channel scale/shift: y = a*raw + b
// ---------------------------------------------------------------------------
__global__ void bn_coeffs(const float* __restrict__ sum, const float* __restrict__ sq,
                          const float* __restrict__ gamma, const float* __restrict__ beta,
                          float* __restrict__ a, float* __restrict__ bshift, int C) {
    int t = blockIdx.x * blockDim.x + threadIdx.x;
    if (t < C) {
        const float invN = 1.0f / (float)NTOT;
        float mean = sum[t] * invN;
        float var  = sq[t] * invN - mean * mean;
        float ai   = gamma[t] * rsqrtf(var + 1e-5f);
        a[t] = ai;
        bshift[t] = beta[t] - ai * mean;
    }
}

// ---------------------------------------------------------------------------
// 5) psi = psi_w . relu(BN(G)+BN(X)) per column; psi stats via LDS tree
// ---------------------------------------------------------------------------
__global__ __launch_bounds__(256)
void psi_dot(const unsigned short* __restrict__ G, const unsigned short* __restrict__ X,
             const float* __restrict__ aG, const float* __restrict__ bG,
             const float* __restrict__ aX, const float* __restrict__ bX,
             const float* __restrict__ psiw,
             float* __restrict__ p_raw, float* __restrict__ psum, float* __restrict__ psq) {
    __shared__ float ls[256], lq[256];
    int tid = threadIdx.x;
    int n = blockIdx.x * 256 + tid;
    float p = 0.0f;
    for (int c = 0; c < F_INT; ++c) {
        float gv = fmaf(aG[c], bf2f(G[(size_t)c * NTOT + n]), bG[c]);
        float xv = fmaf(aX[c], bf2f(X[(size_t)c * NTOT + n]), bX[c]);
        float r = gv + xv;
        r = r > 0.0f ? r : 0.0f;
        p = fmaf(psiw[c], r, p);
    }
    p_raw[n] = p;
    ls[tid] = p; lq[tid] = p * p;
    __syncthreads();
    #pragma unroll
    for (int o = 128; o > 0; o >>= 1) {
        if (tid < o) { ls[tid] += ls[tid + o]; lq[tid] += lq[tid + o]; }
        __syncthreads();
    }
    if (tid == 0) { atomicAdd(psum, ls[0]); atomicAdd(psq, lq[0]); }
}

// ---------------------------------------------------------------------------
// 6) psi-BN + sigmoid -> 65536-entry gate
// ---------------------------------------------------------------------------
__global__ void psi_sigmoid(const float* __restrict__ p_raw,
                            const float* __restrict__ psum, const float* __restrict__ psq,
                            const float* __restrict__ pg, const float* __restrict__ pb,
                            float* __restrict__ gate) {
    int n = blockIdx.x * 256 + threadIdx.x;
    const float invN = 1.0f / (float)NTOT;
    float mean = psum[0] * invN;
    float var  = psq[0] * invN - mean * mean;
    float a    = pg[0] * rsqrtf(var + 1e-5f);
    float bsh  = pb[0] - a * mean;
    float z    = fmaf(a, p_raw[n], bsh);
    gate[n] = 1.0f / (1.0f + __expf(-z));
}

// ---------------------------------------------------------------------------
// 7) out = x * gate  (gate broadcast over channels) — the HBM-bound pass
// ---------------------------------------------------------------------------
__global__ __launch_bounds__(256)
void final_mul(const float* __restrict__ x, const float* __restrict__ gate,
               float* __restrict__ out) {
    size_t i = ((size_t)blockIdx.x * 256 + threadIdx.x) * 4;   // 16M elements
    int hw = (int)(i & (HW - 1));
    int b  = (int)(i >> 20);
    size_t n = ((size_t)b << 12) + hw;
    float4 xv = *(const float4*)(x + i);
    float4 sv = *(const float4*)(gate + n);
    float4 o;
    o.x = xv.x * sv.x; o.y = xv.y * sv.y; o.z = xv.z * sv.z; o.w = xv.w * sv.w;
    *(float4*)(out + i) = o;
}

// ---------------------------------------------------------------------------
extern "C" void kernel_launch(void* const* d_in, const int* in_sizes, int n_in,
                              void* d_out, int out_size, void* d_ws, size_t ws_size,
                              hipStream_t stream) {
    const float* g        = (const float*)d_in[0];
    const float* x        = (const float*)d_in[1];
    const float* wg_w     = (const float*)d_in[2];
    const float* wg_gamma = (const float*)d_in[4];
    const float* wg_beta  = (const float*)d_in[5];
    const float* wx_w     = (const float*)d_in[6];
    const float* wx_gamma = (const float*)d_in[8];
    const float* wx_beta  = (const float*)d_in[9];
    const float* psi_w    = (const float*)d_in[10];
    const float* psi_g    = (const float*)d_in[12];
    const float* psi_be   = (const float*)d_in[13];
    float* out = (float*)d_out;

    char* ws = (char*)d_ws;
    unsigned short* pWg = (unsigned short*)(ws + OFF_PWG);
    unsigned short* pWx = (unsigned short*)(ws + OFF_PWX);
    unsigned short* G   = (unsigned short*)(ws + OFF_G);
    unsigned short* X   = (unsigned short*)(ws + OFF_X);
    float* st = (float*)(ws + OFF_STATS);
    float *sumG = st, *sqG = st + 128, *sumX = st + 256, *sqX = st + 384;
    float *psum = st + 512, *psq = st + 513;
    float *aG = st + 640, *bG = st + 768, *aX = st + 896, *bX = st + 1024;
    float* p_raw = (float*)(ws + OFF_PRAW);
    float* gate  = (float*)(ws + OFF_GATE);

    zero_stats<<<1, 32, 0, stream>>>(psum, 2);
    pack_weights<<<128, 256, 0, stream>>>(wg_w, pWg);
    pack_weights<<<128, 256, 0, stream>>>(wx_w, pWx);
    conv_gemm<<<512, 256, 0, stream>>>(g, pWg, G);
    conv_gemm<<<512, 256, 0, stream>>>(x, pWx, X);
    channel_stats<<<F_INT, 256, 0, stream>>>(G, sumG, sqG);
    channel_stats<<<F_INT, 256, 0, stream>>>(X, sumX, sqX);
    bn_coeffs<<<1, 128, 0, stream>>>(sumG, sqG, wg_gamma, wg_beta, aG, bG, F_INT);
    bn_coeffs<<<1, 128, 0, stream>>>(sumX, sqX, wx_gamma, wx_beta, aX, bX, F_INT);
    psi_dot<<<NTOT / 256, 256, 0, stream>>>(G, X, aG, bG, aX, bX, psi_w, p_raw, psum, psq);
    psi_sigmoid<<<NTOT / 256, 256, 0, stream>>>(p_raw, psum, psq, psi_g, psi_be, gate);
    final_mul<<<16384, 256, 0, stream>>>(x, gate, out);
}
